// ZXGNN_34772055229030
// MI455X (gfx1250) — matrix operand order repaired and checked by reading, not compiled, not run on hardware
//
#include <hip/hip_runtime.h>

#define NN 32768
#define EE 262144
#define DD 256
#define HH 4
#define CC 64
#define GG 32
#define LL 2

typedef __attribute__((ext_vector_type(16))) __bf16 v16bf;
typedef __attribute__((ext_vector_type(8)))  float  v8f;
typedef __attribute__((ext_vector_type(4)))  unsigned int u32x4;
typedef __attribute__((ext_vector_type(8)))  unsigned int u32x8;

union ABu { uint4 u[2]; v16bf v; };

__device__ __forceinline__ float bf2f(unsigned short u) {
    return __uint_as_float(((unsigned int)u) << 16);
}
__device__ __forceinline__ float blo(unsigned int u) {   // low bf16 of a packed dword
    return __uint_as_float(u << 16);
}
__device__ __forceinline__ float bhi(unsigned int u) {   // high bf16 of a packed dword
    return __uint_as_float(u & 0xFFFF0000u);
}
__device__ __forceinline__ unsigned short f2bf(float f) {
    unsigned int x = __float_as_uint(f);
    return (unsigned short)((x + 0x7FFFu + ((x >> 16) & 1u)) >> 16);
}
// monotonic float->uint encoding for atomic max (handles negatives)
__device__ __forceinline__ unsigned int fenc(float f) {
    unsigned int u = __float_as_uint(f);
    return (u >> 31) ? ~u : (u | 0x80000000u);
}
__device__ __forceinline__ float fdec(unsigned int k) {
    return (k >> 31) ? __uint_as_float(k & 0x7FFFFFFFu) : __uint_as_float(~k);
}

__device__ __forceinline__ v8f wmma_bf16(const ABu& a, const ABu& b, v8f c) {
    return __builtin_amdgcn_wmma_f32_16x16x32_bf16(false, a.v, false, b.v, (short)0, c, false, false);
}

// ---------------- utility / prep kernels ----------------

__global__ void k_fill_u32(unsigned int* p, unsigned int v, int n) {
    int t = blockIdx.x * blockDim.x + threadIdx.x;
    if (t < n) p[t] = v;
}

// h = x @ enc_W + enc_b   (x:[N,2], enc_W:[2,D])
__global__ void k_encoder(const float* __restrict__ x, const float* __restrict__ W,
                          const float* __restrict__ b, float* __restrict__ h) {
    int t = blockIdx.x * blockDim.x + threadIdx.x;   // N*D
    int n = t >> 8, d = t & 255;
    h[t] = x[n * 2] * W[d] + x[n * 2 + 1] * W[DD + d] + b[d];
}

__global__ void k_f2bf(const float* __restrict__ s, unsigned short* __restrict__ d, int n) {
    int t = blockIdx.x * blockDim.x + threadIdx.x;
    if (t < n) d[t] = f2bf(s[t]);
}

// Transpose+convert stacked weights [L,D,D] fp32 row-major -> bf16 [L][n][k]
__global__ void k_wtrans(const float* __restrict__ W, unsigned short* __restrict__ Wt) {
    int t = blockIdx.x * blockDim.x + threadIdx.x;   // L*D*D
    int l = t >> 16;                                  // D*D = 65536
    int r = t & 65535;
    int n = r >> 8, k = r & 255;
    Wt[t] = f2bf(W[(l << 16) + (k << 8) + n]);
}

// ---------------- WMMA GEMM: C[N,256] = A[N,256](bf16) x Bt(bf16,[n][k]) + bias ----------------
// block = 128 threads (4 waves); block computes 32 rows x 256 cols; grid.x = N/32.
// The full Bt panel (256x256 bf16 = 128 KB) is staged into LDS once per block by the
// Tensor Data Mover; B fragments are then read from LDS, A fragments from global.
__global__ void __launch_bounds__(128) k_gemm(const unsigned short* __restrict__ A,
                                              const unsigned short* __restrict__ Bt,
                                              const float* __restrict__ bias,
                                              float* __restrict__ Cf,
                                              unsigned short* __restrict__ Cb,
                                              int writeBf) {
    __shared__ __align__(16) unsigned short sB[DD * DD];   // 128 KB, [n][k] bf16

    int lane    = threadIdx.x & 31;
    int wv      = threadIdx.x >> 5;
    int rowBase = blockIdx.x << 5;     // 32 rows per block
    int colBase = wv << 6;             // 64 cols per wave
    int khalf   = lane >> 4;
    int mlo     = lane & 15;

    // ---- TDM: DMA the whole Bt panel into LDS (one wave issues; EXEC ignored by TDM) ----
    if (wv == 0) {
        unsigned int lds = (unsigned int)(size_t)(&sB[0]);
        unsigned long long ga = (unsigned long long)(size_t)Bt;
        u32x4 g0;
        g0[0] = 1u;                                        // count=1 (valid user descriptor)
        g0[1] = lds;                                       // lds_addr
        g0[2] = (unsigned int)(ga & 0xFFFFFFFFu);          // global_addr[31:0]
        g0[3] = (unsigned int)((ga >> 32) & 0x01FFFFFFu)   // global_addr[56:32]
              | (2u << 30);                                // type = 2 ("image")
        u32x8 g1;
        g1[0] = (1u << 16);                 // workgroup_mask=0 (not in cluster), data_size=1 (2B)
        g1[1] = ((unsigned)DD & 0xFFFFu) << 16;            // tensor_dim0 lo16 (=256)
        g1[2] = (((unsigned)DD >> 16) & 0xFFFFu)           // tensor_dim0 hi16
              | (((unsigned)DD & 0xFFFFu) << 16);          // tensor_dim1 lo16 (=256)
        g1[3] = (((unsigned)DD >> 16) & 0xFFFFu)           // tensor_dim1 hi16
              | ((unsigned)DD << 16);                      // tile_dim0 (=256)
        g1[4] = (unsigned)DD;                              // tile_dim1 (=256), tile_dim2=0
        g1[5] = (unsigned)DD;                              // tensor_dim0_stride lo32 (=256)
        g1[6] = 0u;                                        // stride0 hi16, stride1 lo16
        g1[7] = 0u;                                        // stride1 hi32
        u32x4 g2 = {0u, 0u, 0u, 0u};                       // 2D tensor: dims 2/3 unused
        u32x4 g3 = {0u, 0u, 0u, 0u};
        asm volatile("tensor_load_to_lds %0, %1, %2, %3"
                     :: "s"(g0), "s"(g1), "s"(g2), "s"(g3)
                     : "memory");
        __builtin_amdgcn_s_wait_tensorcnt(0);
    }
    __syncthreads();

    const uint4* abase0 = (const uint4*)(A + (((size_t)(rowBase + mlo)) << 8));
    const uint4* abase1 = (const uint4*)(A + (((size_t)(rowBase + 16 + mlo)) << 8));

    v8f acc[8] = {};   // acc[nt*2 + mt]

#pragma unroll
    for (int kt = 0; kt < 8; ++kt) {
        int k0 = kt << 5;
        // A fragments: lane<16 holds K = k0+0..7 and k0+16..23 ; lanes>=16 shift by 8
        ABu a0, a1;
        const uint4* ap0 = abase0 + ((k0 + khalf * 8) >> 3);
        a0.u[0] = ap0[0];
        a0.u[1] = ap0[2];
        const uint4* ap1 = abase1 + ((k0 + khalf * 8) >> 3);
        a1.u[0] = ap1[0];
        a1.u[1] = ap1[2];
#pragma unroll
        for (int nt = 0; nt < 4; ++nt) {
            int n = colBase + (nt << 4) + mlo;
            // B fragment from LDS: lanes 0-15 hold K=k0+0..7 and k0+16..23 ; lanes 16-31 shift by 8
            const uint4* bp = (const uint4*)(&sB[((size_t)n << 8) + k0 + khalf * 8]);
            ABu b;
            b.u[0] = bp[0];
            b.u[1] = bp[2];
            acc[nt * 2 + 0] = wmma_bf16(a0, b, acc[nt * 2 + 0]);
            acc[nt * 2 + 1] = wmma_bf16(a1, b, acc[nt * 2 + 1]);
        }
    }

    // C/D layout: VGPR r, lane: m = base + r + 8*(lane>=16), n = colBase+nt*16+lane%16
#pragma unroll
    for (int nt = 0; nt < 4; ++nt) {
        int nn   = colBase + (nt << 4) + mlo;
        float bs = bias[nn];
#pragma unroll
        for (int mt = 0; mt < 2; ++mt) {
#pragma unroll
            for (int r = 0; r < 8; ++r) {
                int mm     = rowBase + (mt << 4) + r + (khalf << 3);
                float vv   = acc[nt * 2 + mt][r] + bs;
                size_t idx = ((size_t)mm << 8) + nn;
                if (writeBf) Cb[idx] = f2bf(vv);
                else         Cf[idx] = vv;
            }
        }
    }
}

// ---------------- edge attention ----------------

// alpha[e,h] = scale * dot(q[dst,h,:], k[src,h,:] + ea[e]*We[h,:])
__global__ void k_alpha(const unsigned short* __restrict__ qb, const unsigned short* __restrict__ kb,
                        const float* __restrict__ ea, const float* __restrict__ Wev,
                        const int* __restrict__ src, const int* __restrict__ dst,
                        float* __restrict__ alpha) {
    int t = blockIdx.x * blockDim.x + threadIdx.x;   // E*H
    int e = t >> 2, hh = t & 3;
    int s = src[e], d = dst[e];
    float ae = ea[e];
    const uint4* qv = (const uint4*)(qb + ((size_t)d << 8) + (hh << 6));
    const uint4* kv = (const uint4*)(kb + ((size_t)s << 8) + (hh << 6));
    const float* wp = Wev + (hh << 6);
    float acc = 0.f;
#pragma unroll
    for (int i = 0; i < 8; ++i) {
        uint4 qu = qv[i];
        uint4 ku = kv[i];
        unsigned int qa[4] = {qu.x, qu.y, qu.z, qu.w};
        unsigned int ka[4] = {ku.x, ku.y, ku.z, ku.w};
        const float* w8 = wp + i * 8;
#pragma unroll
        for (int j = 0; j < 4; ++j) {
            acc += blo(qa[j]) * (blo(ka[j]) + ae * w8[2 * j]);
            acc += bhi(qa[j]) * (bhi(ka[j]) + ae * w8[2 * j + 1]);
        }
    }
    alpha[t] = acc * 0.125f;   // 1/sqrt(64)
}

__global__ void k_segmax(const float* __restrict__ alpha, const int* __restrict__ dst,
                         unsigned int* __restrict__ mkey) {
    int t = blockIdx.x * blockDim.x + threadIdx.x;
    int e = t >> 2, hh = t & 3;
    atomicMax(&mkey[dst[e] * HH + hh], fenc(alpha[t]));
}

__global__ void k_expden(float* __restrict__ alpha, const int* __restrict__ dst,
                         const unsigned int* __restrict__ mkey, float* __restrict__ den) {
    int t = blockIdx.x * blockDim.x + threadIdx.x;
    int e = t >> 2, hh = t & 3;
    float m  = fdec(mkey[dst[e] * HH + hh]);
    float ex = __expf(alpha[t] - m);
    alpha[t] = ex;
    atomicAdd(&den[dst[e] * HH + hh], ex);
}

__global__ void k_msg(const float* __restrict__ alpha, const float* __restrict__ den,
                      const unsigned short* __restrict__ vb, const float* __restrict__ ea,
                      const float* __restrict__ Wev, const int* __restrict__ src,
                      const int* __restrict__ dst, float* __restrict__ outa) {
    int t = blockIdx.x * blockDim.x + threadIdx.x;
    int e = t >> 2, hh = t & 3;
    int s = src[e], d = dst[e];
    float w  = alpha[t] / den[d * HH + hh];
    float ae = ea[e];
    const uint4* vv = (const uint4*)(vb + ((size_t)s << 8) + (hh << 6));
    const float* wp = Wev + (hh << 6);
    float* op = outa + ((size_t)d << 8) + (hh << 6);
#pragma unroll
    for (int i = 0; i < 8; ++i) {
        uint4 vu = vv[i];
        unsigned int va[4] = {vu.x, vu.y, vu.z, vu.w};
        const float* w8 = wp + i * 8;
#pragma unroll
        for (int j = 0; j < 4; ++j) {
            atomicAdd(&op[i * 8 + 2 * j],     w * (blo(va[j]) + ae * w8[2 * j]));
            atomicAdd(&op[i * 8 + 2 * j + 1], w * (bhi(va[j]) + ae * w8[2 * j + 1]));
        }
    }
}

// ---------------- beta gate + BN + elu ----------------

__global__ void __launch_bounds__(DD) k_beta(const float* __restrict__ outa,
                                             const float* __restrict__ xr,
                                             const float* __restrict__ Wb,
                                             float* __restrict__ h) {
    int n = blockIdx.x, d = threadIdx.x;
    size_t idx = ((size_t)n << 8) + d;
    float o = outa[idx], x = xr[idx];
    float part = o * Wb[d] + x * Wb[DD + d] + (o - x) * Wb[2 * DD + d];
    __shared__ float red[DD];
    red[d] = part;
    __syncthreads();
    for (int s = 128; s > 0; s >>= 1) {
        if (d < s) red[d] += red[d + s];
        __syncthreads();
    }
    float beta = 1.f / (1.f + __expf(-red[0]));
    h[idx] = beta * x + (1.f - beta) * o;
}

__global__ void __launch_bounds__(DD) k_bnstats(const float* __restrict__ h,
                                                float* __restrict__ colsum,
                                                float* __restrict__ colsum2) {
    int d  = threadIdx.x;
    int r0 = blockIdx.x * 128;
    float s = 0.f, s2 = 0.f;
    for (int i = 0; i < 128; ++i) {
        float v = h[((size_t)(r0 + i) << 8) + d];
        s += v; s2 += v * v;
    }
    atomicAdd(&colsum[d], s);
    atomicAdd(&colsum2[d], s2);
}

__global__ void k_bnapply(float* __restrict__ h, const float* __restrict__ colsum,
                          const float* __restrict__ colsum2, const float* __restrict__ gamma,
                          const float* __restrict__ bb) {
    int t = blockIdx.x * blockDim.x + threadIdx.x;   // N*D
    int d = t & 255;
    float mu  = colsum[d] * (1.f / NN);
    float var = colsum2[d] * (1.f / NN) - mu * mu;
    float y = gamma[d] * (h[t] - mu) * rsqrtf(var + 1e-5f) + bb[d];
    h[t] = y > 0.f ? y : (__expf(y) - 1.f);
}

// ---------------- readout ----------------

__global__ void __launch_bounds__(DD) k_gate(const float* __restrict__ h,
                                             const float* __restrict__ gW,
                                             const float* __restrict__ gb,
                                             float* __restrict__ gate) {
    int n = blockIdx.x, d = threadIdx.x;
    __shared__ float red[DD];
    red[d] = h[((size_t)n << 8) + d] * gW[d];
    __syncthreads();
    for (int s = 128; s > 0; s >>= 1) {
        if (d < s) red[d] += red[d + s];
        __syncthreads();
    }
    if (d == 0) gate[n] = red[0] + gb[0];
}

__global__ void k_gmax(const float* __restrict__ gate, const int* __restrict__ batch,
                       unsigned int* __restrict__ gkey) {
    int t = blockIdx.x * blockDim.x + threadIdx.x;   // N
    atomicMax(&gkey[batch[t]], fenc(gate[t]));
}

__global__ void k_gexp(float* __restrict__ gate, const int* __restrict__ batch,
                       const unsigned int* __restrict__ gkey, float* __restrict__ gden) {
    int t = blockIdx.x * blockDim.x + threadIdx.x;   // N
    int b = batch[t];
    float ex = __expf(gate[t] - fdec(gkey[b]));
    gate[t] = ex;
    atomicAdd(&gden[b], ex);
}

__global__ void k_pool(const float* __restrict__ h, const float* __restrict__ gate,
                       const float* __restrict__ gden, const int* __restrict__ batch,
                       float* __restrict__ hG) {
    int t = blockIdx.x * blockDim.x + threadIdx.x;   // N*D
    int n = t >> 8, d = t & 255;
    int b = batch[n];
    float a = gate[n] / gden[b];
    atomicAdd(&hG[(b << 8) + d], a * h[t]);
}

__global__ void __launch_bounds__(DD) k_final(const float* __restrict__ hG,
                                              const float* __restrict__ dW,
                                              const float* __restrict__ db,
                                              const float* __restrict__ ob,
                                              float* __restrict__ out) {
    int g = blockIdx.x, d = threadIdx.x;
    __shared__ float red[DD];
    red[d] = hG[(g << 8) + d] * dW[d];
    __syncthreads();
    for (int s = 128; s > 0; s >>= 1) {
        if (d < s) red[d] += red[d + s];
        __syncthreads();
    }
    if (d == 0) out[g] = red[0] + db[0] + ob[0];
}

// ---------------- host ----------------

extern "C" void kernel_launch(void* const* d_in, const int* in_sizes, int n_in,
                              void* d_out, int out_size, void* d_ws, size_t ws_size,
                              hipStream_t stream) {
    (void)in_sizes; (void)n_in; (void)out_size; (void)ws_size;
    const float* x    = (const float*)d_in[0];
    const int*   ei   = (const int*)d_in[1];
    const float* ea   = (const float*)d_in[2];
    const int*   batch= (const int*)d_in[3];
    const float* encW = (const float*)d_in[4];
    const float* encb = (const float*)d_in[5];
    const float* Wq   = (const float*)d_in[6];
    const float* bq   = (const float*)d_in[7];
    const float* Wk   = (const float*)d_in[8];
    const float* bk   = (const float*)d_in[9];
    const float* Wv   = (const float*)d_in[10];
    const float* bv   = (const float*)d_in[11];
    const float* We   = (const float*)d_in[12];
    const float* Ws   = (const float*)d_in[13];
    const float* bs   = (const float*)d_in[14];
    const float* Wb   = (const float*)d_in[15];
    const float* gam  = (const float*)d_in[16];
    const float* bet  = (const float*)d_in[17];
    const float* gW   = (const float*)d_in[18];
    const float* gb   = (const float*)d_in[19];
    const float* dW   = (const float*)d_in[20];
    const float* db   = (const float*)d_in[21];
    const float* ob   = (const float*)d_in[22];
    const int* src = ei;
    const int* dst = ei + EE;

    char* ws = (char*)d_ws;
    size_t off = 0;
    auto alloc = [&](size_t bytes) -> char* {
        char* p = ws + off;
        off += (bytes + 255) & ~(size_t)255;
        return p;
    };
    float*          h    = (float*)         alloc((size_t)NN * DD * 4);
    unsigned short* hb   = (unsigned short*)alloc((size_t)NN * DD * 2);
    unsigned short* qb   = (unsigned short*)alloc((size_t)NN * DD * 2);
    unsigned short* kb   = (unsigned short*)alloc((size_t)NN * DD * 2);
    unsigned short* vb   = (unsigned short*)alloc((size_t)NN * DD * 2);
    float*          xr   = (float*)         alloc((size_t)NN * DD * 4);
    float*          outa = (float*)         alloc((size_t)NN * DD * 4);
    float*          alpha= (float*)         alloc((size_t)EE * HH * 4);
    unsigned int*   mkey = (unsigned int*)  alloc((size_t)NN * HH * 4);
    float*          den  = (float*)         alloc((size_t)NN * HH * 4);
    unsigned short* WqT  = (unsigned short*)alloc((size_t)LL * DD * DD * 2);
    unsigned short* WkT  = (unsigned short*)alloc((size_t)LL * DD * DD * 2);
    unsigned short* WvT  = (unsigned short*)alloc((size_t)LL * DD * DD * 2);
    unsigned short* WsT  = (unsigned short*)alloc((size_t)LL * DD * DD * 2);
    float*          colsum = (float*)       alloc((size_t)DD * 4);     // colsum2 follows contiguously
    float*          colsum2= (float*)       alloc((size_t)DD * 4);
    float*          gate = (float*)         alloc((size_t)NN * 4);
    unsigned int*   gkey = (unsigned int*)  alloc((size_t)GG * 4);
    float*          gden = (float*)         alloc((size_t)GG * 4);
    float*          hG   = (float*)         alloc((size_t)GG * DD * 4);
    (void)colsum2;

    const int TPB = 256;

    // weight prep (bf16, transposed)
    k_wtrans<<<LL * DD * DD / TPB, TPB, 0, stream>>>(Wq, WqT);
    k_wtrans<<<LL * DD * DD / TPB, TPB, 0, stream>>>(Wk, WkT);
    k_wtrans<<<LL * DD * DD / TPB, TPB, 0, stream>>>(Wv, WvT);
    k_wtrans<<<LL * DD * DD / TPB, TPB, 0, stream>>>(Ws, WsT);

    // encoder
    k_encoder<<<NN * DD / TPB, TPB, 0, stream>>>(x, encW, encb, h);

    for (int l = 0; l < LL; ++l) {
        k_f2bf<<<NN * DD / TPB, TPB, 0, stream>>>(h, hb, NN * DD);

        k_gemm<<<NN / 32, 128, 0, stream>>>(hb, WqT + l * DD * DD, bq + l * DD, (float*)nullptr, qb, 1);
        k_gemm<<<NN / 32, 128, 0, stream>>>(hb, WkT + l * DD * DD, bk + l * DD, (float*)nullptr, kb, 1);
        k_gemm<<<NN / 32, 128, 0, stream>>>(hb, WvT + l * DD * DD, bv + l * DD, (float*)nullptr, vb, 1);
        k_gemm<<<NN / 32, 128, 0, stream>>>(hb, WsT + l * DD * DD, bs + l * DD, xr, (unsigned short*)nullptr, 0);

        const float* Wev = We + l * DD;

        k_fill_u32<<<(NN * HH + TPB - 1) / TPB, TPB, 0, stream>>>(mkey, 0x007FFFFFu, NN * HH); // enc(-inf)
        k_fill_u32<<<(NN * HH + TPB - 1) / TPB, TPB, 0, stream>>>((unsigned int*)den, 0u, NN * HH);
        k_fill_u32<<<(NN * DD + TPB - 1) / TPB, TPB, 0, stream>>>((unsigned int*)outa, 0u, NN * DD);
        k_fill_u32<<<(2 * DD + TPB - 1) / TPB, TPB, 0, stream>>>((unsigned int*)colsum, 0u, 2 * DD);

        k_alpha <<<EE * HH / TPB, TPB, 0, stream>>>(qb, kb, ea, Wev, src, dst, alpha);
        k_segmax<<<EE * HH / TPB, TPB, 0, stream>>>(alpha, dst, mkey);
        k_expden<<<EE * HH / TPB, TPB, 0, stream>>>(alpha, dst, mkey, den);
        k_msg   <<<EE * HH / TPB, TPB, 0, stream>>>(alpha, den, vb, ea, Wev, src, dst, outa);

        k_beta   <<<NN, DD, 0, stream>>>(outa, xr, Wb + l * 3 * DD, h);
        k_bnstats<<<NN / 128, DD, 0, stream>>>(h, colsum, colsum + DD);
        k_bnapply<<<NN * DD / TPB, TPB, 0, stream>>>(h, colsum, colsum + DD, gam + l * DD, bet + l * DD);
    }

    // readout
    k_fill_u32<<<1, TPB, 0, stream>>>(gkey, 0x007FFFFFu, GG);
    k_fill_u32<<<1, TPB, 0, stream>>>((unsigned int*)gden, 0u, GG);
    k_fill_u32<<<GG * DD / TPB, TPB, 0, stream>>>((unsigned int*)hG, 0u, GG * DD);

    k_gate <<<NN, DD, 0, stream>>>(h, gW, gb, gate);
    k_gmax <<<NN / TPB, TPB, 0, stream>>>(gate, batch, gkey);
    k_gexp <<<NN / TPB, TPB, 0, stream>>>(gate, batch, gkey, gden);
    k_pool <<<NN * DD / TPB, TPB, 0, stream>>>(h, gate, gden, batch, hG);
    k_final<<<GG, DD, 0, stream>>>(hG, dW, db, ob, (float*)d_out);
}
